// Flashback_12043088298505
// MI455X (gfx1250) — compile-verified, matrix operand order checked by reading, not run on hardware
//
#include <hip/hip_runtime.h>
#include <hip/hip_bf16.h>
#include <math.h>

// ---------------------------------------------------------------------------
// Flashback RNN + weighted pooling + big FC, CDNA5 (gfx1250) wave32 WMMA.
//   S=200, B=1024, H=128, V=100000, U=10000, K=2H=256
// ---------------------------------------------------------------------------

typedef __bf16 bf16_t;
typedef __attribute__((ext_vector_type(16))) __bf16 v16bf;
typedef __attribute__((ext_vector_type(8)))  float  v8f;

#define S_LEN 200
#define BATCH 1024
#define HDIM  128
#define VOCAB 100000
#define KDIM  256
#define A_STRIDE 264   // 16x256 A tile padded: 528B row stride -> 4-bank rotate, conflict-free b128 loads

// Branch-free tanh: single V_TANH_F32 trans op if the builtin exists,
// otherwise exp2-based (1 trans + rcp), no exec divergence on the RNN
// critical path (libm tanhf emits branchy cndmask chains).
__device__ __forceinline__ float fast_tanh(float x) {
#if __has_builtin(__builtin_amdgcn_tanhf)
  return __builtin_amdgcn_tanhf(x);
#else
  const float xc = fminf(fmaxf(x, -9.0f), 9.0f);
  const float e  = __builtin_amdgcn_exp2f(xc * 2.8853900817779268f); // 2*log2(e)
  return (e - 1.f) * __builtin_amdgcn_rcpf(e + 1.f);
#endif
}

// A/B fragment loader for bf16 16x16x32 WMMA from a row-major bf16 row pointer.
// lane covers row/col (lane&15); low half-wave holds K {kk..kk+7, kk+16..kk+23},
// high half-wave holds K {kk+8..kk+15, kk+24..kk+31}  (per CDNA5 ISA 7.12.2).
__device__ __forceinline__ v16bf load_frag_bf16(const bf16_t* rowbase, int kk, int hi) {
  const bf16_t* p = rowbase + kk + hi * 8;
  union { v16bf v; float4 q[2]; } u;
  u.q[0] = *(const float4*)(p);        // K = kk+hi*8 .. +7      (16B)
  u.q[1] = *(const float4*)(p + 16);   // K = kk+16+hi*8 .. +7   (16B)
  return u.v;
}

// Same fragment pattern but converting from an fp32 row on the fly.
__device__ __forceinline__ v16bf load_frag_f32(const float* rowbase, int kk, int hi) {
  const float* p = rowbase + kk + hi * 8;
  v16bf f;
#pragma unroll
  for (int j = 0; j < 8; ++j) {
    f[j]     = (bf16_t)p[j];
    f[j + 8] = (bf16_t)p[j + 16];
  }
  return f;
}

// Pack 8 fp32 -> 8 bf16 and store as one b128 (dst must be 16B aligned).
__device__ __forceinline__ void cvt_store8(bf16_t* dst, const float* src) {
  union { bf16_t h[8]; float4 q; } pk;
#pragma unroll
  for (int j = 0; j < 8; ++j) pk.h[j] = (bf16_t)src[j];
  *(float4*)dst = pk.q;
}

// ---------------------------------------------------------------------------
// Kernel 1: fused embedding-gather + 200-step tanh RNN + ragged weighted pool.
// Grid: B/16 blocks x 256 threads (8 waves). Wave w owns H-slice n0 = w*16.
// Recurrent weights live in registers as resident B-fragments for all steps.
// All pooling weights precomputed once into LDS (keeps the sequential loop
// free of libm / divergence). Writes Z[b,0:128]=out_w, Z[b,128:256]=p_u (bf16).
// ---------------------------------------------------------------------------
__global__ __launch_bounds__(256) void rnn_pool_kernel(
    const int* __restrict__ x, const float* __restrict__ t,
    const float* __restrict__ sc, const float* __restrict__ h0,
    const int* __restrict__ active_user, const int* __restrict__ length,
    const float* __restrict__ W_enc, const float* __restrict__ W_user,
    const float* __restrict__ W_ih, const float* __restrict__ W_hh,
    const float* __restrict__ b_ih, const float* __restrict__ b_hh,
    bf16_t* __restrict__ Z)
{
  __shared__ bf16_t A2[16 * A_STRIDE];   // [16 rows][K=256]: 0..127 = x_emb, 128..255 = h
  __shared__ float  w_all[S_LEN * 16];   // pooling weights w[s][row]
  __shared__ float  den_lds[16];
  __shared__ float  tl_lds[16], sx_lds[16], sy_lds[16];
  __shared__ int    len_lds[16];

  const int tid  = threadIdx.x;
  const int lane = tid & 31;
  const int wid  = tid >> 5;
  const int hi   = lane >> 4;
  const int lc   = lane & 15;
  const int b0   = blockIdx.x * 16;
  const int nn   = wid * 16 + lc;          // output H column this lane owns

  // Resident B-fragments: rows nn of [W_ih | W_hh] (kb 0..3 -> W_ih, 4..7 -> W_hh)
  v16bf bfr[8];
#pragma unroll
  for (int kb = 0; kb < 8; ++kb) {
    const float* wsrc = (kb < 4 ? W_ih : W_hh) + (size_t)nn * HDIM;
    bfr[kb] = load_frag_f32(wsrc, (kb & 3) * 32, hi);
  }

  // Initial hidden state -> A2 cols 128..255 ; last-event data -> LDS
  {
    const int row = tid >> 4, col = (tid & 15) * 8;
    cvt_store8(A2 + row * A_STRIDE + 128 + col,
               h0 + (size_t)(b0 + row) * HDIM + col);
  }
  if (tid < 16) {
    const int b = b0 + tid;
    const int Lb = length[b];
    const size_t li = (size_t)(Lb - 1) * BATCH + b;
    len_lds[tid] = Lb;
    tl_lds[tid]  = t[li];
    sx_lds[tid]  = sc[li * 2 + 0];
    sy_lds[tid]  = sc[li * 2 + 1];
  }
  __syncthreads();

  // One-time precompute of all 200x16 pooling weights (parallel, fast intrinsics)
  for (int e = tid; e < S_LEN * 16; e += 256) {
    const int s = e >> 4, r = e & 15;
    float w = 0.f;
    if (s < len_lds[r]) {
      const size_t si = (size_t)s * BATCH + (b0 + r);
      const float dt = tl_lds[r] - t[si];
      const float dx = sx_lds[r] - sc[si * 2 + 0];
      const float dy = sy_lds[r] - sc[si * 2 + 1];
      const float ft = (__cosf(dt * 7.27220521664304e-05f) + 1.f) * 0.5f *
                       __expf(-dt * 1.15740740740741e-06f);
      const float fs = __expf(-sqrtf(dx * dx + dy * dy) * 1000.f);
      w = ft * fs + 1e-10f;
    }
    w_all[e] = w;
  }
  __syncthreads();

  const float bias = b_ih[nn] + b_hh[nn];
  v8f num = {};
  const int grow = tid >> 4;
  const int gcol = (tid & 15) * 8;

  for (int s = 0; s < S_LEN; ++s) {
    // ---- phase A: gather this step's embeddings into A2 cols 0..127
    {
      const int token = x[(size_t)s * BATCH + b0 + grow];
      cvt_store8(A2 + grow * A_STRIDE + gcol,
                 W_enc + (size_t)token * HDIM + gcol);
    }
    __syncthreads();

    // ---- phase B: [16,256] x [256, 16-slice] via 8 WMMAs; tanh; pool
    v8f acc = {};
    const bf16_t* arow = A2 + lc * A_STRIDE;
#pragma unroll
    for (int kb = 0; kb < 8; ++kb) {
      const v16bf a = load_frag_bf16(arow, kb * 32, hi);
      acc = __builtin_amdgcn_wmma_f32_16x16x32_bf16(
          false, a, false, bfr[kb], (short)0, acc, false, false);
    }
    float hn[8];
#pragma unroll
    for (int i = 0; i < 8; ++i) {
      const float hv = fast_tanh(acc[i] + bias);   // C VGPR i -> row M = i + 8*hi
      hn[i] = hv;
      num[i] += w_all[s * 16 + i + 8 * hi] * hv;
    }
    __syncthreads();

    // ---- phase C: write new hidden state (A2 cols 128..255) for next step
#pragma unroll
    for (int i = 0; i < 8; ++i)
      A2[(i + 8 * hi) * A_STRIDE + 128 + nn] = (bf16_t)hn[i];
  }

  // Denominators (one-time 16-thread reduction over the weight table)
  if (tid < 16) {
    float d = 0.f;
    for (int s = 0; s < S_LEN; ++s) d += w_all[s * 16 + tid];
    den_lds[tid] = d;
  }
  __syncthreads();

  // out_w -> Z[:, 0:128] (bf16)
#pragma unroll
  for (int i = 0; i < 8; ++i) {
    const int m = i + 8 * hi;
    Z[(size_t)(b0 + m) * KDIM + nn] = (bf16_t)(num[i] / den_lds[m]);
  }
  // user embedding -> Z[:, 128:256]
  {
    const int u = active_user[b0 + grow];
    cvt_store8(Z + (size_t)(b0 + grow) * KDIM + 128 + gcol,
               W_user + (size_t)u * HDIM + gcol);
  }
}

// ---------------------------------------------------------------------------
// Kernel 2: out[b,v] = Z[b,:] . W_fc[v,:] + b_fc[v]
// [1024,256] x [256,100000]. Each wave: one 16-col V tile x two 16-row B tiles
// (shared B-fragments), K=256 -> 16 WMMAs. W_fc converted fp32->bf16 on the fly.
// Memory-bound: 102 MB W_fc reads (L2-resident across 32 b-blocks) + 410 MB out.
// ---------------------------------------------------------------------------
__global__ __launch_bounds__(256) void fc_kernel(
    const bf16_t* __restrict__ Z, const float* __restrict__ W_fc,
    const float* __restrict__ b_fc, float* __restrict__ out)
{
  const int tid   = threadIdx.x;
  const int lane  = tid & 31;
  const int wid   = tid >> 5;
  const int hi    = lane >> 4;
  const int lc    = lane & 15;
  const int ntile = blockIdx.x * 8 + wid;
  if (ntile >= VOCAB / 16) return;
  const int v0 = ntile * 16;
  const int b0 = blockIdx.y * 32;

  // Resident A fragments for two M tiles (rows b0+lc and b0+16+lc of Z)
  const bf16_t* zr0 = Z + (size_t)(b0 + lc) * KDIM;
  const bf16_t* zr1 = Z + (size_t)(b0 + 16 + lc) * KDIM;
  v16bf a0[8], a1[8];
#pragma unroll
  for (int kb = 0; kb < 8; ++kb) {
    a0[kb] = load_frag_bf16(zr0, kb * 32, hi);
    a1[kb] = load_frag_bf16(zr1, kb * 32, hi);
  }

  const float* wrow = W_fc + (size_t)(v0 + lc) * KDIM;  // lane owns vocab row v0+lc
  v8f c0 = {}, c1 = {};
#pragma unroll
  for (int kb = 0; kb < 8; ++kb) {
    const v16bf bb = load_frag_f32(wrow, kb * 32, hi);
    c0 = __builtin_amdgcn_wmma_f32_16x16x32_bf16(
        false, a0[kb], false, bb, (short)0, c0, false, false);
    c1 = __builtin_amdgcn_wmma_f32_16x16x32_bf16(
        false, a1[kb], false, bb, (short)0, c1, false, false);
  }

  const float bias = b_fc[v0 + lc];
#pragma unroll
  for (int i = 0; i < 8; ++i) {
    out[(size_t)(b0 + i + 8 * hi) * VOCAB + v0 + lc]      = c0[i] + bias;
    out[(size_t)(b0 + 16 + i + 8 * hi) * VOCAB + v0 + lc] = c1[i] + bias;
  }
}

// ---------------------------------------------------------------------------
extern "C" void kernel_launch(void* const* d_in, const int* in_sizes, int n_in,
                              void* d_out, int out_size, void* d_ws, size_t ws_size,
                              hipStream_t stream) {
  const int*   x      = (const int*)  d_in[0];
  const float* t      = (const float*)d_in[1];
  const float* sc     = (const float*)d_in[2];
  // d_in[3] (y_t), d_in[4] (y_s) unused by the reference computation
  const float* h0     = (const float*)d_in[5];
  const int*   au     = (const int*)  d_in[6];
  const int*   len    = (const int*)  d_in[7];
  const float* W_enc  = (const float*)d_in[8];
  const float* W_user = (const float*)d_in[9];
  const float* W_ih   = (const float*)d_in[10];
  const float* W_hh   = (const float*)d_in[11];
  const float* b_ih   = (const float*)d_in[12];
  const float* b_hh   = (const float*)d_in[13];
  const float* W_fc   = (const float*)d_in[14];
  const float* b_fc   = (const float*)d_in[15];
  float*  out = (float*)d_out;
  bf16_t* Z   = (bf16_t*)d_ws;  // [B, 256] bf16 = 512 KB

  rnn_pool_kernel<<<BATCH / 16, 256, 0, stream>>>(
      x, t, sc, h0, au, len, W_enc, W_user, W_ih, W_hh, b_ih, b_hh, Z);

  dim3 g2((VOCAB / 16 + 7) / 8, BATCH / 32);
  fc_kernel<<<g2, 256, 0, stream>>>(Z, W_fc, b_fc, out);
}
